// Encoder_74938589381000
// MI455X (gfx1250) — compile-verified, matrix operand order checked by reading, not run on hardware
//
#include <hip/hip_runtime.h>

typedef __attribute__((ext_vector_type(16))) _Float16 v16h;
typedef __attribute__((ext_vector_type(8)))  _Float16 v8h;
typedef __attribute__((ext_vector_type(8)))  float    v8f;
typedef int v4i __attribute__((vector_size(16)));

#define AS1 __attribute__((address_space(1)))
#define AS3 __attribute__((address_space(3)))

// Unaligned-capable 16B vector load wrapper.
struct __attribute__((packed)) V8HU { v8h v; };

#if __has_builtin(__builtin_amdgcn_global_load_async_to_lds_b128) && \
    __has_builtin(__builtin_amdgcn_s_wait_asynccnt)
#define USE_ASYNC 1
#else
#define USE_ASYNC 0
#endif

// ---------------------------------------------------------------------------
// Fused Conv1d as implicit GEMM on WMMA f32_16x16x32_f16, double-buffered:
// prefetch K-chunk i+1 into LDS buffer (i+1)&1 (async DMA where eligible)
// while WMMAs consume chunk i. One barrier per chunk.
//   Y[b,co,t] = bias[co] + sum_{ci,k} W[k,co,ci] * act(X[b,ci,t*stride-pad+k*dil])
// Weights pre-repacked to [Kk][Cout][Cin] (contiguous along Cin).
// Block: 256 threads = 8 waves; tile 64(M=Cout) x 128(N=B*Tout); wave -> 32x32.
// ---------------------------------------------------------------------------
struct ConvP {
  const _Float16* X;     // [B, Cin, Tin] f16
  const _Float16* W;     // [Kk, Cout, Cin] f16 (repacked)
  const float*    bias;  // [Cout] or null
  const _Float16* R;     // residual [B, Cout, Tout] or null
  _Float16*       Yh;    // f16 out or null
  float*          Yf;    // f32 out or null
  int B, Cin, Cout, Tin, Tout, Kk, stride, dil, pad, preRelu;
};

__global__ __launch_bounds__(256, 2)
void conv1d_wmma(ConvP p) {
  __shared__ _Float16 sA[2][64][40];    // [buf][m][kk]  weights
  __shared__ _Float16 sB[2][32][136];   // [buf][kk][n]  im2col

  const int tid  = threadIdx.x;
  const int lane = tid & 31;
  const int wave = tid >> 5;
  const int wm   = wave & 1;         // 0..1 : M subtile
  const int wn   = wave >> 1;        // 0..3 : N subtile
  const int NT   = p.B * p.Tout;
  const int n0   = blockIdx.x * 128;
  const int m0   = blockIdx.y * 64;

  // A-tile unit: thread -> (m, 8 consecutive kk). 64x32 halfs = 256 units.
  const int am  = tid >> 2;
  const int aks = (tid & 3) * 8;
  // B-tile units: thread -> two units (kk0, kk0+16) x (8 consecutive n).
  const int bn    = (tid & 15) * 8;
  const int kk0   = tid >> 4;            // 0..15
  const int col0  = n0 + bn;
  const int colOk = (col0 + 7 < NT);
  int bb = 0, t0 = 0;
  if (col0 < NT) { bb = col0 / p.Tout; t0 = col0 - bb * p.Tout; }
  const int tinBase = t0 * p.stride - p.pad;
  const int xrow0   = bb * p.Cin * p.Tin;

  const int nCc     = (p.Cin + 31) >> 5;   // Cin chunks of 32
  const int nChunks = p.Kk * nCc;

  // ---- stage one K-chunk into LDS buffer `buf` ----
  auto stage = [&](int chunk, int buf) {
    const int k  = chunk / nCc;
    const int c0 = (chunk - k * nCc) * 32;
    const bool aFull = (m0 + 64 <= p.Cout) && (c0 + 32 <= p.Cin);
    // A tile (weights, contiguous along Cin)
    {
      const int gm   = m0 + am;
      const int aoff = (k * p.Cout + gm) * p.Cin + (c0 + aks);
#if USE_ASYNC
      if (aFull && ((p.Cin & 7) == 0)) {
        __builtin_amdgcn_global_load_async_to_lds_b128(
            (AS1 v4i*)(p.W + aoff), (AS3 v4i*)&sA[buf][am][aks], 0, 0);
      } else
#endif
      if (aFull) {
        *(v8h*)&sA[buf][am][aks] = ((const V8HU*)(p.W + aoff))->v;
      } else {
        v8h v;
        #pragma unroll
        for (int e = 0; e < 8; ++e) {
          const int gc = c0 + aks + e;
          const int ok = (gm < p.Cout) & (gc < p.Cin);
          const int ix = ok ? (aoff + e) : 0;
          const _Float16 t = p.W[ix];           // always-executed, clamped
          v[e] = ok ? t : (_Float16)0.f;
        }
        *(v8h*)&sA[buf][am][aks] = v;
      }
    }
    // B tile (im2col, contiguous along t)
    const int tin0 = tinBase + k * p.dil;
    #pragma unroll
    for (int u = 0; u < 2; ++u) {
      const int kk   = kk0 + u * 16;
      const int gc   = c0 + kk;
      const int xrow = xrow0 + gc * p.Tin;
      const bool okC = colOk && (gc < p.Cin);
      const bool f1  = okC && (p.stride == 1) && (tin0 >= 0) && (tin0 + 7 < p.Tin);
#if USE_ASYNC
      if (f1 && !p.preRelu && ((tin0 & 7) == 0)) {
        __builtin_amdgcn_global_load_async_to_lds_b128(
            (AS1 v4i*)(p.X + xrow + tin0), (AS3 v4i*)&sB[buf][kk][bn], 0, 0);
      } else
#endif
      {
        v8h v;
        if (f1) {
          v = ((const V8HU*)(p.X + xrow + tin0))->v;
        } else if (okC && (p.stride == 2) && (tin0 >= 0) && (tin0 + 15 < p.Tin)) {
          const v8h lo = ((const V8HU*)(p.X + xrow + tin0))->v;
          const v8h hi = ((const V8HU*)(p.X + xrow + tin0 + 8))->v;
          #pragma unroll
          for (int e = 0; e < 4; ++e) { v[e] = lo[2 * e]; v[e + 4] = hi[2 * e]; }
        } else {
          #pragma unroll
          for (int e = 0; e < 8; ++e) {
            const int tin = tin0 + e * p.stride;
            const int ok  = (col0 + e < NT) & (gc < p.Cin) & (tin >= 0) & (tin < p.Tin);
            const int ix  = ok ? (xrow + tin) : 0;
            const _Float16 t = p.X[ix];         // always-executed, clamped
            v[e] = ok ? t : (_Float16)0.f;
          }
        }
        if (p.preRelu) {
          #pragma unroll
          for (int e = 0; e < 8; ++e)
            v[e] = ((float)v[e] > 0.f) ? v[e] : (_Float16)0.f;
        }
        *(v8h*)&sB[buf][kk][bn] = v;
      }
    }
  };

  v8f acc[2][2] = {};
  const int r15 = lane & 15;
  const int kb  = (lane >> 4) << 3;

  stage(0, 0);
  for (int ch = 0; ch < nChunks; ++ch) {
    const int buf = ch & 1;
    // Split point: stage(ch) writes (incl. async DMA) complete and visible;
    // also separates compute(ch-1) reads from stage(ch+1) writes to buf^1.
#if USE_ASYNC
    __builtin_amdgcn_s_wait_asynccnt(0);
#endif
    __syncthreads();
    if (ch + 1 < nChunks) stage(ch + 1, buf ^ 1);   // overlaps with WMMAs below

    // ---- fragments per CDNA5 16-bit WMMA layout ----
    // lanes 0-15: K in {0..7, 16..23}; lanes 16-31: K in {8..15, 24..31}
    v16h afr[2], bfr[2];
    #pragma unroll
    for (int h = 0; h < 2; ++h) {
      const int mr = wm * 32 + h * 16 + r15;
      const int nc = wn * 32 + h * 16 + r15;
      #pragma unroll
      for (int e = 0; e < 16; ++e) {
        const int kk = kb + (e & 7) + ((e >> 3) << 4);
        afr[h][e] = sA[buf][mr][kk];
        bfr[h][e] = sB[buf][kk][nc];
      }
    }
    #pragma unroll
    for (int i = 0; i < 2; ++i)
      #pragma unroll
      for (int j = 0; j < 2; ++j)
        acc[i][j] = __builtin_amdgcn_wmma_f32_16x16x32_f16(
            false, afr[i], false, bfr[j], (short)0, acc[i][j], false, false);
  }

  // ---- epilogue: D layout = VGPR r -> M = base + (lane>=16 ? 8:0) + r, N = lane&15
  const int rhi = (lane >> 4) << 3;
  #pragma unroll
  for (int i = 0; i < 2; ++i) {
    const int rowb = m0 + wm * 32 + i * 16 + rhi;
    #pragma unroll
    for (int j = 0; j < 2; ++j) {
      const int col = n0 + wn * 32 + j * 16 + r15;
      if (col >= NT) continue;
      const int b = col / p.Tout, t = col - b * p.Tout;
      #pragma unroll
      for (int r = 0; r < 8; ++r) {
        const int row = rowb + r;
        if (row >= p.Cout) break;
        float v = acc[i][j][r];
        if (p.bias) v += p.bias[row];
        const int oidx = (b * p.Cout + row) * p.Tout + t;
        if (p.R)  v += (float)p.R[oidx];
        if (p.Yh) p.Yh[oidx] = (_Float16)v;
        if (p.Yf) p.Yf[oidx] = v;
      }
    }
  }
}

// f32 -> f16 elementwise
__global__ void cvt_f32_f16(const float* __restrict__ s, _Float16* __restrict__ d, int n) {
  const int i = blockIdx.x * blockDim.x + threadIdx.x;
  if (i < n) d[i] = (_Float16)s[i];
}

// f32 [Cout,Cin,K] -> f16 repacked [K,Cout,Cin]
__global__ void cvt_repack(const float* __restrict__ s, _Float16* __restrict__ d,
                           int Cout, int Cin, int Kk) {
  const int i = blockIdx.x * blockDim.x + threadIdx.x;
  const int total = Cout * Cin * Kk;
  if (i >= total) return;
  const int k  = i / (Cout * Cin);
  const int r  = i - k * Cout * Cin;
  const int co = r / Cin;
  const int ci = r - co * Cin;
  d[i] = (_Float16)s[(co * Cin + ci) * Kk + k];
}

// GCN symmetric-norm mix, batch 0 only: rows 0,1 -> 0.5*(h0+h1)
__global__ void gcn_mix(_Float16* __restrict__ hu, _Float16* __restrict__ hl, int n) {
  const int i = blockIdx.x * blockDim.x + threadIdx.x;
  if (i < n) {
    const float m = 0.5f * ((float)hu[i] + (float)hl[i]);
    hu[i] = (_Float16)m;
    hl[i] = (_Float16)m;
  }
}

// ---------------------------------------------------------------------------
extern "C" void kernel_launch(void* const* d_in, const int* in_sizes, int n_in,
                              void* d_out, int out_size, void* d_ws, size_t ws_size,
                              hipStream_t stream) {
  (void)in_sizes; (void)n_in; (void)out_size; (void)ws_size;
  constexpr int BS = 64, T = 256, UF = 156, LF = 107, WD = 512;

  int idx = 0;
  const float* ux    = (const float*)d_in[idx++];
  const float* lx    = (const float*)d_in[idx++];
  const float* W_up  = (const float*)d_in[idx++];
  const float* b_up  = (const float*)d_in[idx++];
  const float* W_gcn = (const float*)d_in[idx++];
  const float* b_gcn = (const float*)d_in[idx++];

  struct BranchF { const float *wd[3], *bd[3], *w1[3][3], *b1[3][3], *w2[3][3], *b2[3][3], *wf, *bf; };
  auto parse = [&](BranchF& br) {
    for (int blk = 0; blk < 3; ++blk) {
      br.wd[blk] = (const float*)d_in[idx++];
      br.bd[blk] = (const float*)d_in[idx++];
      for (int j = 0; j < 3; ++j) {
        br.w1[blk][j] = (const float*)d_in[idx++];
        br.b1[blk][j] = (const float*)d_in[idx++];
        br.w2[blk][j] = (const float*)d_in[idx++];
        br.b2[blk][j] = (const float*)d_in[idx++];
      }
    }
    br.wf = (const float*)d_in[idx++];
    br.bf = (const float*)d_in[idx++];
  };
  BranchF ub, lb;
  parse(ub);
  parse(lb);

  // Workspace bump allocator (256B aligned slots).
  size_t off = 0;
  auto arena = [&](size_t elems) -> _Float16* {
    _Float16* pp = (_Float16*)((char*)d_ws + off);
    off += (elems * sizeof(_Float16) + 255) & ~(size_t)255;
    return pp;
  };
  // Convert + repack weights [Cout,Cin,K] f32 -> [K,Cout,Cin] f16.
  auto toHW = [&](const float* s, int Cout, int Cin, int Kk) -> const _Float16* {
    const size_t n = (size_t)Cout * Cin * Kk;
    _Float16* d = arena(n);
    cvt_repack<<<dim3((unsigned)((n + 255) / 256)), dim3(256), 0, stream>>>(s, d, Cout, Cin, Kk);
    return d;
  };

  // Activation arenas.
  _Float16* ux16  = arena((size_t)BS * UF * T);
  _Float16* lx16  = arena((size_t)BS * LF * T);
  _Float16* lup16 = arena((size_t)BS * UF * T);
  _Float16* hu    = arena((size_t)BS * WD * T);
  _Float16* hl    = arena((size_t)BS * WD * T);
  _Float16* sb0   = arena((size_t)BS * WD * (T / 2));
  _Float16* sb1   = arena((size_t)BS * WD * (T / 2));
  _Float16* sb2   = arena((size_t)BS * WD * (T / 2));
  _Float16* sb[3] = {sb0, sb1, sb2};

  const _Float16* Wup16 = toHW(W_up, UF, LF, 3);
  const _Float16* Wg16  = toHW(W_gcn, WD, UF, 1);
  struct BranchH { const _Float16 *wd[3], *w1[3][3], *w2[3][3], *wf; };
  auto cvtBranch = [&](const BranchF& f, BranchH& h) {
    for (int blk = 0; blk < 3; ++blk) {
      h.wd[blk] = toHW(f.wd[blk], WD, WD, 4);
      for (int j = 0; j < 3; ++j) {
        h.w1[blk][j] = toHW(f.w1[blk][j], WD, WD, 3);
        h.w2[blk][j] = toHW(f.w2[blk][j], WD, WD, 1);
      }
    }
    h.wf = toHW(f.wf, WD, WD, 3);
  };
  BranchH ubh, lbh;
  cvtBranch(ub, ubh);
  cvtBranch(lb, lbh);

  auto runConv = [&](const _Float16* X, const _Float16* Wt, const float* bias,
                     const _Float16* R, _Float16* Yh, float* Yf,
                     int Cin, int Cout, int Tin, int Tout,
                     int Kk, int stride, int dil, int pad, int preRelu) {
    ConvP p;
    p.X = X; p.W = Wt; p.bias = bias; p.R = R; p.Yh = Yh; p.Yf = Yf;
    p.B = BS; p.Cin = Cin; p.Cout = Cout; p.Tin = Tin; p.Tout = Tout;
    p.Kk = Kk; p.stride = stride; p.dil = dil; p.pad = pad; p.preRelu = preRelu;
    dim3 g((BS * Tout + 127) / 128, (Cout + 63) / 64);
    conv1d_wmma<<<g, dim3(256), 0, stream>>>(p);
  };

  // ---- Front end ----
  const int nUx = BS * UF * T, nLx = BS * LF * T;
  cvt_f32_f16<<<(nUx + 255) / 256, 256, 0, stream>>>(ux, ux16, nUx);
  cvt_f32_f16<<<(nLx + 255) / 256, 256, 0, stream>>>(lx, lx16, nLx);
  // lbody upsample: Conv1d(107 -> 156, k=3, pad=1)
  runConv(lx16, Wup16, b_up, nullptr, lup16, nullptr, LF, UF, T, T, 3, 1, 1, 1, 0);
  // GCN linear (k=1 conv, 156 -> 512) + bias
  runConv(ux16,  Wg16, b_gcn, nullptr, hu, nullptr, UF, WD, T, T, 1, 1, 1, 0, 0);
  runConv(lup16, Wg16, b_gcn, nullptr, hl, nullptr, UF, WD, T, T, 1, 1, 1, 0, 0);
  // Mix the two global nodes of batch 0 only.
  gcn_mix<<<(WD * T + 255) / 256, 256, 0, stream>>>(hu, hl, WD * T);

  // ---- Branches ----
  auto runBranch = [&](const BranchF& f, const BranchH& h, _Float16* hin, float* out) {
    const int dils[3] = {9, 3, 1};
    _Float16* cur = hin;
    int ci = -1;
    int Tc = T;
    for (int blk = 0; blk < 3; ++blk) {
      const int d = (ci != 0) ? 0 : 1;
      runConv(cur, h.wd[blk], f.bd[blk], nullptr, sb[d], nullptr,
              WD, WD, Tc, Tc / 2, 4, 2, 1, 1, 0);
      ci = d; cur = sb[d]; Tc /= 2;
      for (int j = 0; j < 3; ++j) {
        int t1 = -1, t2 = -1;
        for (int q = 0; q < 3; ++q)
          if (q != ci) { if (t1 < 0) t1 = q; else t2 = q; }
        // h1 = conv(relu(x), w1, k=3, dil)
        runConv(cur, h.w1[blk][j], f.b1[blk][j], nullptr, sb[t1], nullptr,
                WD, WD, Tc, Tc, 3, 1, dils[j], dils[j], 1);
        // y = x + conv(relu(h1), w2, k=1)
        runConv(sb[t1], h.w2[blk][j], f.b2[blk][j], cur, sb[t2], nullptr,
                WD, WD, Tc, Tc, 1, 1, 1, 0, 1);
        ci = t2; cur = sb[t2];
      }
    }
    // final Conv1d(512 -> 512, k=3, pad=1), f32 output straight to d_out
    runConv(cur, h.wf, f.bf, nullptr, nullptr, out, WD, WD, Tc, Tc, 3, 1, 1, 1, 0);
  };

  float* outF = (float*)d_out;
  runBranch(ub, ubh, hu, outF);
  runBranch(lb, lbh, hl, outF + (size_t)BS * WD * 32);
}
